// Attn_AR_24824910971026
// MI455X (gfx1250) — compile-verified, matrix operand order checked by reading, not run on hardware
//
#include <hip/hip_runtime.h>
#include <hip/hip_bf16.h>

// Problem dims (fixed by the reference)
#define BB 256
#define SS 512
#define HH 256
#define DD 512
#define SH (SS*HH)

typedef __attribute__((ext_vector_type(16))) __bf16 v16bf;
typedef __attribute__((ext_vector_type(8)))  float  v8f;
typedef int v4i __attribute__((vector_size(16)));   // matches builtin param pointee

union FragU   { v16bf v; uint4 q[2]; };
union Pack8bf { uint4 q; __bf16 h[8]; };
union Acc8    { v8f v; float f[8]; float4 f4[2]; };

// CDNA5 async global->LDS copy path (ASYNCcnt-tracked), with safe fallback.
#if defined(__has_builtin)
#if __has_builtin(__builtin_amdgcn_global_load_async_to_lds_b128) && \
    __has_builtin(__builtin_amdgcn_s_wait_asynccnt)
#define HAVE_ASYNC_LDS 1
#endif
#endif

#if defined(HAVE_ASYNC_LDS)
#define WAIT_ASYNC(n) __builtin_amdgcn_s_wait_asynccnt(n)
#else
#define WAIT_ASYNC(n) do { } while (0)
#endif

static __device__ inline void copy16_g2lds(const __bf16* gsrc, __bf16* ldst) {
#if defined(HAVE_ASYNC_LDS)
  __builtin_amdgcn_global_load_async_to_lds_b128(
      (__attribute__((address_space(1))) v4i*)gsrc,
      (__attribute__((address_space(3))) v4i*)ldst,
      0, 0);
#else
  *reinterpret_cast<uint4*>(ldst) = *reinterpret_cast<const uint4*>(gsrc);
#endif
}

// Load one 16x32 bf16 WMMA fragment from global memory.
// Works for both A (row = M) and B given as Bt[N,K] row-major (row = N).
static __device__ inline v16bf load_frag_bf16(const __bf16* __restrict__ base,
                                              long ld, int k0, int lane) {
  const int row = lane & 15;
  const int hi  = lane >> 4;
  const __bf16* p = base + (long)row * ld + (k0 + hi * 8);
  FragU u;
  u.q[0] = *reinterpret_cast<const uint4*>(p);
  u.q[1] = *reinterpret_cast<const uint4*>(p + 16);
  return u.v;
}

// Same, reading from an LDS-resident strip (lowered to ds_load_b128).
static __device__ inline v16bf load_frag_lds(const __bf16* base, int ld, int k0, int lane) {
  const int row = lane & 15;
  const int hi  = lane >> 4;
  const __bf16* p = base + row * ld + (k0 + hi * 8);
  FragU u;
  u.q[0] = *reinterpret_cast<const uint4*>(p);
  u.q[1] = *reinterpret_cast<const uint4*>(p + 16);
  return u.v;
}

static __device__ inline v8f wmma_bf16(v16bf a, v16bf b, v8f c) {
  return __builtin_amdgcn_wmma_f32_16x16x32_bf16(false, a, false, b,
                                                 (short)0, c, false, false);
}

// ---------------- f32 -> bf16 convert (vectorized) ----------------
__global__ void k_f32_to_bf16_v4(const float* __restrict__ src,
                                 __bf16* __restrict__ dst, long n4) {
  long i = (long)blockIdx.x * blockDim.x + threadIdx.x;
  if (i >= n4) return;
  float4 f = reinterpret_cast<const float4*>(src)[i];
  union { uint2 u; __bf16 h[4]; } o;
  o.h[0] = (__bf16)f.x; o.h[1] = (__bf16)f.y;
  o.h[2] = (__bf16)f.z; o.h[3] = (__bf16)f.w;
  reinterpret_cast<uint2*>(dst)[i] = o.u;
}

// ---------------- QKV projection: [B*S,D] x W[H,D]^T + bias ----------------
// Block: 256 threads = 8 waves. The 16xD A-strip is staged once in LDS
// (async copy), then each wave computes two 16x16 n-tiles (all of H covered).
// z=0 -> kT [B,H,S], z=1 -> qT [B,H,S], z=2 -> v [B,S,H]
#define QLDA (DD + 8)   // 520: 1040B row stride -> rows 4 banks apart, no conflicts
__global__ void k_qkv(const __bf16* __restrict__ xb, const __bf16* __restrict__ wb,
                      const float* __restrict__ kb, const float* __restrict__ qb,
                      const float* __restrict__ vb,
                      __bf16* __restrict__ kT, __bf16* __restrict__ qT,
                      __bf16* __restrict__ vv) {
  __shared__ __bf16 As[16 * QLDA];
  const int tid   = threadIdx.x;
  const int lane  = tid & 31;
  const int wave  = tid >> 5;      // 0..7
  const int mtile = blockIdx.x;    // over B*S/16
  const int z     = blockIdx.z;    // k/q/v

  // Stage 16 x 512 bf16 strip of x: 1024 16B chunks, 4 per thread.
  const __bf16* Ag = xb + (long)mtile * 16 * DD;
  #pragma unroll
  for (int i = 0; i < 4; ++i) {
    int c   = tid + i * 256;
    int row = c >> 6;
    int col = (c & 63) * 8;
    copy16_g2lds(Ag + (long)row * DD + col, As + row * QLDA + col);
  }
  WAIT_ASYNC(0);
  __syncthreads();

  const __bf16* Wz = wb + (long)z * HH * DD;
  const __bf16* W0 = Wz + (long)(wave * 2 + 0) * 16 * DD;
  const __bf16* W1 = Wz + (long)(wave * 2 + 1) * 16 * DD;
  v8f acc0 = {0.f,0.f,0.f,0.f,0.f,0.f,0.f,0.f};
  v8f acc1 = {0.f,0.f,0.f,0.f,0.f,0.f,0.f,0.f};
  for (int s = 0; s < DD / 32; ++s) {
    v16bf a  = load_frag_lds(As, QLDA, s * 32, lane);
    v16bf b0 = load_frag_bf16(W0, DD, s * 32, lane);
    v16bf b1 = load_frag_bf16(W1, DD, s * 32, lane);
    acc0 = wmma_bf16(a, b0, acc0);
    acc1 = wmma_bf16(a, b1, acc1);
  }

  const int col = lane & 15, hi = lane >> 4;
  const int sg0 = mtile * 16;
  const int b   = sg0 / SS;
  const int s0  = sg0 % SS;        // tile never straddles a batch (512 % 16 == 0)
  #pragma unroll
  for (int u = 0; u < 2; ++u) {
    Acc8 c; c.v = (u == 0) ? acc0 : acc1;
    const int ntile = wave * 2 + u;
    const int h = ntile * 16 + col;
    const float bias = (z == 0) ? kb[h] : ((z == 1) ? qb[h] : vb[h]);
    if (z < 2) {
      __bf16* dstT = (z == 0) ? kT : qT;
      Pack8bf o;
      #pragma unroll
      for (int r = 0; r < 8; ++r) o.h[r] = (__bf16)(c.f[r] + bias);
      // rows m = r + 8*hi map to consecutive s -> one 16B store
      *reinterpret_cast<uint4*>(dstT + ((long)b * HH + h) * SS + s0 + hi * 8) = o.q;
    } else {
      #pragma unroll
      for (int r = 0; r < 8; ++r) {
        int s = s0 + 8 * hi + r;
        vv[((long)b * SS + s) * HH + h] = (__bf16)(c.f[r] + bias);
      }
    }
  }
}

// ---------------- scores = scale * K^T Q, stored transposed [B,G,H] ---------
__global__ void k_scores(const __bf16* __restrict__ kT, const __bf16* __restrict__ qT,
                         float* __restrict__ scT) {
  const int lane  = threadIdx.x & 31;
  const int mtile = blockIdx.x;   // h tiles
  const int ntile = blockIdx.y;   // g tiles
  const int b     = blockIdx.z;
  const __bf16* A  = kT + ((long)b * HH + mtile * 16) * SS;
  const __bf16* Bq = qT + ((long)b * HH + ntile * 16) * SS;
  v8f acc = {0.f,0.f,0.f,0.f,0.f,0.f,0.f,0.f};
  for (int s = 0; s < SS / 32; ++s) {
    v16bf a  = load_frag_bf16(A,  SS, s * 32, lane);
    v16bf bb = load_frag_bf16(Bq, SS, s * 32, lane);
    acc = wmma_bf16(a, bb, acc);
  }
  const float scale = 0.0625f;    // 256^-0.5
  Acc8 c; c.v = acc * scale;
  const int col = lane & 15, hi = lane >> 4;
  const int g  = ntile * 16 + col;
  const int h0 = mtile * 16 + 8 * hi;   // 8 consecutive h per lane
  float* p = scT + ((long)b * HH + g) * HH + h0;
  reinterpret_cast<float4*>(p)[0] = c.f4[0];
  reinterpret_cast<float4*>(p)[1] = c.f4[1];
}

// ---------------- softmax over h (contiguous rows of scT) -> bf16 wT --------
__global__ void k_softmax(const float* __restrict__ scT, __bf16* __restrict__ wT,
                          long nrows) {
  long gid  = (long)blockIdx.x * blockDim.x + threadIdx.x;
  long row  = gid >> 5;
  int  lane = threadIdx.x & 31;
  if (row >= nrows) return;
  const float4* p = reinterpret_cast<const float4*>(scT + row * HH);
  float4 a = p[lane * 2], bq = p[lane * 2 + 1];
  float x[8] = {a.x, a.y, a.z, a.w, bq.x, bq.y, bq.z, bq.w};
  float m = x[0];
  #pragma unroll
  for (int j = 1; j < 8; ++j) m = fmaxf(m, x[j]);
  for (int off = 16; off; off >>= 1) m = fmaxf(m, __shfl_xor(m, off, 32));
  float s = 0.f;
  #pragma unroll
  for (int j = 0; j < 8; ++j) { x[j] = __expf(x[j] - m); s += x[j]; }
  for (int off = 16; off; off >>= 1) s += __shfl_xor(s, off, 32);
  float inv = 1.f / s;
  Pack8bf o;
  #pragma unroll
  for (int j = 0; j < 8; ++j) o.h[j] = (__bf16)(x[j] * inv);
  *reinterpret_cast<uint4*>(wT + row * HH + lane * 8) = o.q;
}

// ---------------- out1[b,s,g] = V x W + attn_bias ---------------------------
__global__ void k_attn(const __bf16* __restrict__ vv, const __bf16* __restrict__ wT,
                       const float* __restrict__ attn_bias, float* __restrict__ out1) {
  const int lane  = threadIdx.x & 31;
  const int mtile = blockIdx.x;   // s tiles
  const int ntile = blockIdx.y;   // g tiles
  const int b     = blockIdx.z;
  const __bf16* A  = vv + ((long)b * SS + mtile * 16) * HH;
  const __bf16* Bw = wT + ((long)b * HH + ntile * 16) * HH;
  v8f acc = {0.f,0.f,0.f,0.f,0.f,0.f,0.f,0.f};
  for (int s = 0; s < HH / 32; ++s) {
    v16bf a  = load_frag_bf16(A,  HH, s * 32, lane);
    v16bf bb = load_frag_bf16(Bw, HH, s * 32, lane);
    acc = wmma_bf16(a, bb, acc);
  }
  Acc8 c; c.v = acc;
  const int col = lane & 15, hi = lane >> 4;
  const int g = ntile * 16 + col;
  #pragma unroll
  for (int r = 0; r < 8; ++r) {
    int s = mtile * 16 + 8 * hi + r;
    out1[((long)b * SS + s) * HH + g] = c.f[r] + attn_bias[s * HH + g];
  }
}

// ---------------- LayerNorm(H) + exact GELU -> bf16 [B, S*H] ----------------
__global__ void k_ln_gelu(const float* __restrict__ out1, const float* __restrict__ ln_g,
                          const float* __restrict__ ln_b, __bf16* __restrict__ attnb,
                          long nrows) {
  long gid  = (long)blockIdx.x * blockDim.x + threadIdx.x;
  long row  = gid >> 5;
  int  lane = threadIdx.x & 31;
  if (row >= nrows) return;
  const float4* p = reinterpret_cast<const float4*>(out1 + row * HH);
  float4 a = p[lane * 2], bq = p[lane * 2 + 1];
  float x[8] = {a.x, a.y, a.z, a.w, bq.x, bq.y, bq.z, bq.w};
  float s = 0.f, s2 = 0.f;
  #pragma unroll
  for (int j = 0; j < 8; ++j) { s += x[j]; s2 += x[j] * x[j]; }
  for (int off = 16; off; off >>= 1) { s += __shfl_xor(s, off, 32); s2 += __shfl_xor(s2, off, 32); }
  float mu  = s * (1.f / HH);
  float var = s2 * (1.f / HH) - mu * mu;
  float rs  = rsqrtf(var + 1e-5f);
  Pack8bf o;
  #pragma unroll
  for (int j = 0; j < 8; ++j) {
    int h = lane * 8 + j;
    float y = (x[j] - mu) * rs * ln_g[h] + ln_b[h];
    y = 0.5f * y * (1.f + erff(y * 0.70710678118654752f));   // exact GELU
    o.h[j] = (__bf16)y;
  }
  *reinterpret_cast<uint4*>(attnb + row * HH + lane * 8) = o.q;
}

// ---------------- init out[b,d] = out_b[d] ----------------------------------
__global__ void k_init_out(float* __restrict__ out, const float* __restrict__ out_b) {
  int i = blockIdx.x * blockDim.x + threadIdx.x;   // B*D threads
  out[i] = out_b[i % DD];
}

// ---------------- final GEMM [B,SH] x out_w[D,SH]^T, split-K + atomics ------
// Block: 256 threads = 8 waves, one 16-row A strip shared by all waves
// (each wave owns one 16-col d-tile). K chunk of 2048 runs as 4 async
// double-buffered 512-wide LDS stages: copy(next) overlaps compute(cur).
#define KSPLIT 64
#define KCHUNK (SH / KSPLIT)    // 2048
#define FSTK   512              // K columns per LDS stage
#define FNST   (KCHUNK / FSTK)  // 4 stages
#define FLDA   (FSTK + 8)       // 520 -> conflict-free row stride
__global__ void k_final(const __bf16* __restrict__ attnb, const __bf16* __restrict__ owb,
                        float* __restrict__ out) {
  __shared__ __bf16 As[2][16 * FLDA];
  const int tid   = threadIdx.x;
  const int lane  = tid & 31;
  const int wave  = tid >> 5;     // 0..7
  const int mtile = blockIdx.x;   // B/16
  const int dgrp  = blockIdx.y;   // D/(16*8)
  const int kc    = blockIdx.z;   // KSPLIT
  const long kbase = (long)kc * KCHUNK;
  const __bf16* Ag = attnb + (long)mtile * 16 * SH + kbase;
  const int d0     = (dgrp * 8 + wave) * 16;
  const __bf16* Bw = owb + (long)d0 * SH + kbase;

  // stage st -> buffer st&1 : 1024 16B chunks, 4 per thread
  auto issue_stage = [&](int st) {
    __bf16* dst = As[st & 1];
    #pragma unroll
    for (int i = 0; i < 4; ++i) {
      int c   = tid + i * 256;
      int row = c >> 6;
      int col = (c & 63) * 8;
      copy16_g2lds(Ag + (long)row * SH + st * FSTK + col, dst + row * FLDA + col);
    }
  };

  v8f acc = {0.f,0.f,0.f,0.f,0.f,0.f,0.f,0.f};
  issue_stage(0);
  for (int st = 0; st < FNST; ++st) {
    if (st + 1 < FNST) {
      issue_stage(st + 1);                 // overwrites buf last read in st-1
      __builtin_prefetch(Bw + (long)(lane & 15) * SH + (st + 1) * FSTK, 0, 0);
      WAIT_ASYNC(4);                       // my 4 copies of stage st are done
    } else {
      WAIT_ASYNC(0);
    }
    __syncthreads();                       // everyone's stage-st data visible
    const __bf16* buf = As[st & 1];
    for (int s = 0; s < FSTK / 32; ++s) {
      v16bf a  = load_frag_lds(buf, FLDA, s * 32, lane);
      v16bf bb = load_frag_bf16(Bw, SH, st * FSTK + s * 32, lane);
      acc = wmma_bf16(a, bb, acc);
    }
    __syncthreads();                       // done reading buf before reuse
  }

  Acc8 c; c.v = acc;
  const int col = lane & 15, hi = lane >> 4;
  const int d = d0 + col;
  #pragma unroll
  for (int r = 0; r < 8; ++r) {
    int b = mtile * 16 + 8 * hi + r;
    atomicAdd(&out[(long)b * DD + d], c.f[r]);
  }
}

extern "C" void kernel_launch(void* const* d_in, const int* in_sizes, int n_in,
                              void* d_out, int out_size, void* d_ws, size_t ws_size,
                              hipStream_t stream) {
  (void)in_sizes; (void)n_in; (void)out_size; (void)ws_size;
  const float* x         = (const float*)d_in[0];
  const float* k_w       = (const float*)d_in[1];
  const float* k_b       = (const float*)d_in[2];
  const float* q_w       = (const float*)d_in[3];
  const float* q_b       = (const float*)d_in[4];
  const float* v_w       = (const float*)d_in[5];
  const float* v_b       = (const float*)d_in[6];
  const float* attn_bias = (const float*)d_in[7];
  const float* ln_g      = (const float*)d_in[8];
  const float* ln_b      = (const float*)d_in[9];
  const float* out_w     = (const float*)d_in[10];
  const float* out_b     = (const float*)d_in[11];
  float* out = (float*)d_out;

  char* ws = (char*)d_ws;
  size_t off = 0;
  __bf16* xb  = (__bf16*)(ws + off); off += (size_t)BB * SS * DD * 2;   // x in bf16
  __bf16* owb = (__bf16*)(ws + off); off += (size_t)DD * SH * 2;        // out_w bf16
  __bf16* wb  = (__bf16*)(ws + off); off += (size_t)3 * HH * DD * 2;    // k/q/v weights
  size_t off_kT = off;
  __bf16* kT  = (__bf16*)(ws + off); off += (size_t)BB * HH * SS * 2;   // K^T
  __bf16* qT  = (__bf16*)(ws + off); off += (size_t)BB * HH * SS * 2;   // Q^T
  __bf16* vv  = (__bf16*)(ws + off); off += (size_t)BB * SS * HH * 2;   // V
  size_t off_sc = off;
  float*  scT = (float*)(ws + off);  off += (size_t)BB * HH * HH * 4;   // scores^T
  __bf16* wT  = (__bf16*)(ws + off); off += (size_t)BB * HH * HH * 2;   // softmax^T
  float*  out1  = (float*)(ws + off_kT);   // alias kT+qT (dead after k_scores)
  __bf16* attnb = (__bf16*)(ws + off_sc);  // alias scT   (dead after k_softmax)

  // f32 -> bf16 converts
  {
    long n4 = (long)BB * SS * DD / 4;
    k_f32_to_bf16_v4<<<dim3((unsigned)((n4 + 255) / 256)), 256, 0, stream>>>(x, xb, n4);
    n4 = (long)DD * SH / 4;
    k_f32_to_bf16_v4<<<dim3((unsigned)((n4 + 255) / 256)), 256, 0, stream>>>(out_w, owb, n4);
    n4 = (long)HH * DD / 4;
    unsigned g = (unsigned)((n4 + 255) / 256);
    k_f32_to_bf16_v4<<<g, 256, 0, stream>>>(k_w, wb,               n4);
    k_f32_to_bf16_v4<<<g, 256, 0, stream>>>(q_w, wb + 1 * HH * DD, n4);
    k_f32_to_bf16_v4<<<g, 256, 0, stream>>>(v_w, wb + 2 * HH * DD, n4);
  }

  k_qkv    <<<dim3(BB * SS / 16, 1, 3),      256, 0, stream>>>(xb, wb, k_b, q_b, v_b, kT, qT, vv);
  k_scores <<<dim3(HH / 16, HH / 16, BB),     32, 0, stream>>>(kT, qT, scT);
  k_softmax<<<dim3(BB * HH / 8),             256, 0, stream>>>(scT, wT, (long)BB * HH);
  k_attn   <<<dim3(SS / 16, HH / 16, BB),     32, 0, stream>>>(vv, wT, attn_bias, out1);
  k_ln_gelu<<<dim3(BB * SS / 8),             256, 0, stream>>>(out1, ln_g, ln_b, attnb, (long)BB * SS);
  k_init_out<<<dim3(BB * DD / 256),          256, 0, stream>>>(out, out_b);
  k_final  <<<dim3(BB / 16, DD / (16 * 8), KSPLIT), 256, 0, stream>>>(attnb, owb, out);
}
